// Gramba_26912265076977
// MI455X (gfx1250) — matrix operand order repaired and checked by reading, not compiled, not run on hardware
//
#include <hip/hip_runtime.h>
#include <math.h>

// ---------------------------------------------------------------------------
// minGRU block for MI455X (gfx1250, wave32, WMMA, async-to-LDS staging).
//   pre-pass: x, W1, W2, Wg, Wo converted fp32 -> bf16 once (halves L2 traffic)
//   x_in  = silu(x @ W1^T + b1)            bf16 WMMA GEMM, bf16 out
//   x_skip= silu(x @ W2^T + b2)            bf16 WMMA GEMM, f32 out
//   gh    = x_in @ Wg^T + bg   -> epilogue computes log_coeffs / log_z / log_g
//   scan  : Heinsen log-space recurrence over T, fused y = bf16(x_skip + h)
//   out   = y @ Wo^T + bo                  bf16 WMMA GEMM, f32 out
// GEMM staging uses GLOBAL_LOAD_ASYNC_TO_LDS_B128 (ASYNCcnt) with LDS
// double-buffering so global fetch of stage i+1 overlaps WMMA of stage i.
// ---------------------------------------------------------------------------

typedef __bf16        v16bf  __attribute__((ext_vector_type(16)));
typedef float         v8f    __attribute__((ext_vector_type(8)));
typedef float         f32x4  __attribute__((ext_vector_type(4)));
typedef unsigned int  u32x4  __attribute__((ext_vector_type(4)));
typedef unsigned int  u32x2  __attribute__((ext_vector_type(2)));

union Frag16 { v16bf v; u32x4 q[2]; };
union BfPack4 { __bf16 h[4]; u32x2 q; };

__device__ __forceinline__ float siluf(float v) {
    return v / (1.0f + __expf(-v));
}
__device__ __forceinline__ float softplusf(float v) {
    return fmaxf(v, 0.0f) + log1pf(__expf(-fabsf(v)));
}
__device__ __forceinline__ float log_gf(float v) {          // log(g(v))
    return (v >= 0.0f) ? __logf(v + 0.5f) : -softplusf(-v);
}

// ---------------------------------------------------------------------------
// fp32 -> bf16 bulk conversion (one-time per launch, n % 4 == 0)
// ---------------------------------------------------------------------------
__global__ __launch_bounds__(256)
void cvt_f32_bf16_kernel(const float* __restrict__ src, __bf16* __restrict__ dst,
                         long n)
{
    const long i = ((long)blockIdx.x * 256 + threadIdx.x) * 4;
    if (i < n) {
        const f32x4 v = *(const f32x4*)(src + i);
        BfPack4 pk;
        pk.h[0] = (__bf16)v[0]; pk.h[1] = (__bf16)v[1];
        pk.h[2] = (__bf16)v[2]; pk.h[3] = (__bf16)v[3];
        *(u32x2*)(dst + i) = pk.q;
    }
}

// ---------------------------------------------------------------------------
// Tiled bf16 WMMA GEMM:  C[M,N] = epilogue( A[M,K] * W[N,K]^T + bias[N] )
//   EPI 0 : silu -> bf16 store to C0        (ldc = N)
//   EPI 1 : silu -> f32 store to C0         (ldc = N)
//   EPI 2 : gate epilogue. halfN = N/2.
//           n <  halfN : C0[m,n] = -softplus(v) ; C1[m,n] = -softplus(-v)
//           n >= halfN : C2[m,n-halfN] = log_g(v)
//   EPI 3 : f32 store to C0                 (ldc = N)
// Block: 256 threads (8 waves), tile 128x128, K-stage 64, double-buffered LDS.
// Wave grid 4(M) x 2(N): each wave 32x64 -> 2x4 wmma f32 accumulators.
// ---------------------------------------------------------------------------
constexpr int BM = 128, BN = 128, BK = 64;
constexpr int PITCH = BK + 8;                // 72 bf16 = 144B pitch (16B mult)
constexpr int STAGE = (BM + BN) * PITCH;     // elements per LDS stage buffer

// Issue one stage (A tile 128xBK + W tile 128xBK) of async 16B lane-copies.
// 8 instructions per wave per stage -> ASYNCcnt += 8.
__device__ __forceinline__ void async_stage(const __bf16* __restrict__ gA,
                                            const __bf16* __restrict__ gW,
                                            long K, long mBlock, long nBlock,
                                            int k0, unsigned ldsA, unsigned ldsW,
                                            int tid)
{
    const int c    = tid & 7;                 // 16B chunk within row (8 bf16)
    const int row0 = tid >> 3;                // 0..31
#pragma unroll
    for (int p = 0; p < 4; ++p) {
        const int r = row0 + p * 32;
        const unsigned long long ga =
            (unsigned long long)(gA + (mBlock + r) * K + k0 + c * 8);
        const unsigned la = ldsA + (unsigned)(r * PITCH + c * 8) * 2u;
        asm volatile("global_load_async_to_lds_b128 %0, %1, off"
                     :: "v"(la), "v"(ga) : "memory");
    }
#pragma unroll
    for (int p = 0; p < 4; ++p) {
        const int r = row0 + p * 32;
        const unsigned long long ga =
            (unsigned long long)(gW + (nBlock + r) * K + k0 + c * 8);
        const unsigned lw = ldsW + (unsigned)(r * PITCH + c * 8) * 2u;
        asm volatile("global_load_async_to_lds_b128 %0, %1, off"
                     :: "v"(lw), "v"(ga) : "memory");
    }
}

template <int EPI>
__global__ __launch_bounds__(256)
void gemm_wmma_kernel(const __bf16* __restrict__ A,
                      const __bf16* __restrict__ W,
                      const float* __restrict__ bias,
                      void* __restrict__ C0, void* __restrict__ C1,
                      void* __restrict__ C2,
                      int M, int N, int K)
{
    __shared__ __bf16 smem[2 * STAGE];       // double-buffered A|W stages

    const int tid  = threadIdx.x;
    const int lane = tid & 31;
    const int wid  = tid >> 5;
    const int wm   = (wid & 3) * 32;          // wave M offset inside block tile
    const int wn   = (wid >> 2) * 64;         // wave N offset inside block tile
    const int hh   = lane >> 4;               // lane half (0/1)
    const int l15  = lane & 15;

    const long mBlock = (long)blockIdx.y * BM;
    const long nBlock = (long)blockIdx.x * BN;
    const long Kl     = K;

    // LDS aperture keeps the wave-relative byte offset in addr[31:0];
    // the async-to-LDS instruction adds LDS_BASE itself.
    const unsigned ldsBase = (unsigned)(unsigned long long)(const void*)smem;

    v8f acc[2][4];
#pragma unroll
    for (int mi = 0; mi < 2; ++mi)
#pragma unroll
        for (int ni = 0; ni < 4; ++ni)
            acc[mi][ni] = (v8f){0.f,0.f,0.f,0.f,0.f,0.f,0.f,0.f};

    // prologue: stage 0 into buffer 0
    async_stage(A, W, Kl, mBlock, nBlock, 0,
                ldsBase, ldsBase + (unsigned)(BM * PITCH) * 2u, tid);

    int cur = 0;
    for (int k0 = 0; k0 < K; k0 += BK) {
        const int nxt = cur ^ 1;
        if (k0 + BK < K) {
            // issue stage i+1 into the other buffer, then retire stage i only
            async_stage(A, W, Kl, mBlock, nBlock, k0 + BK,
                        ldsBase + (unsigned)(nxt * STAGE) * 2u,
                        ldsBase + (unsigned)(nxt * STAGE + BM * PITCH) * 2u, tid);
            asm volatile("s_wait_asynccnt 0x8" ::: "memory");
        } else {
            asm volatile("s_wait_asynccnt 0x0" ::: "memory");
        }
        __syncthreads();

        const int aOff = cur * STAGE;
        const int wOff = aOff + BM * PITCH;

#pragma unroll
        for (int kk = 0; kk < BK; kk += 32) {
            Frag16 a[2], b[4];
#pragma unroll
            for (int mi = 0; mi < 2; ++mi) {
                // A 16x32 bf16: lanes 0-15 -> K 0..7,16..23 ; 16-31 -> K 8..15,24..31
                const int row = wm + mi * 16 + l15;
                a[mi].q[0] = *(const u32x4*)&smem[aOff + row * PITCH + kk + hh * 8];
                a[mi].q[1] = *(const u32x4*)&smem[aOff + row * PITCH + kk + 16 + hh * 8];
            }
#pragma unroll
            for (int ni = 0; ni < 4; ++ni) {
                // B 32x16 bf16: lanes 0-15 -> K 0..7,16..23 ; 16-31 -> K 8..15,24..31
                const int row = wn + ni * 16 + l15;
                b[ni].q[0] = *(const u32x4*)&smem[wOff + row * PITCH + kk + hh * 8];
                b[ni].q[1] = *(const u32x4*)&smem[wOff + row * PITCH + kk + 16 + hh * 8];
            }
#pragma unroll
            for (int mi = 0; mi < 2; ++mi)
#pragma unroll
                for (int ni = 0; ni < 4; ++ni)
                    acc[mi][ni] = __builtin_amdgcn_wmma_f32_16x16x32_bf16(
                        false, a[mi].v, false, b[ni].v,
                        (short)0, acc[mi][ni], false, false);
        }
        __syncthreads();   // all waves done reading buf[cur] before its reuse
        cur = nxt;
    }

    // ---- epilogue: C/D layout lane<16: N=lane, VGPR r -> M=r ; lane>=16: M=8+r
    const long halfN = (long)N >> 1;
#pragma unroll
    for (int mi = 0; mi < 2; ++mi) {
#pragma unroll
        for (int ni = 0; ni < 4; ++ni) {
            const long nb = nBlock + wn + ni * 16 + l15;
            const long mb = mBlock + wm + mi * 16 + hh * 8;
            const float bv = bias[nb];
#pragma unroll
            for (int r = 0; r < 8; ++r) {
                const long m = mb + r;
                const float v = acc[mi][ni][r] + bv;
                if (EPI == 0) {
                    ((__bf16*)C0)[m * (long)N + nb] = (__bf16)siluf(v);
                } else if (EPI == 1) {
                    ((float*)C0)[m * (long)N + nb] = siluf(v);
                } else if (EPI == 2) {
                    if (nb < halfN) {
                        ((float*)C0)[m * halfN + nb] = -softplusf(v);   // log(1-z)
                        ((float*)C1)[m * halfN + nb] = -softplusf(-v);  // log z
                    } else {
                        ((float*)C2)[m * halfN + (nb - halfN)] = log_gf(v);
                    }
                } else {
                    ((float*)C0)[m * (long)N + nb] = v;
                }
            }
        }
    }
}

// ---------------------------------------------------------------------------
// Heinsen log-space scan over T, fused with y = bf16(x_skip + h).
// One thread per (b,e) channel; consecutive lanes -> consecutive e (coalesced).
// ---------------------------------------------------------------------------
__global__ __launch_bounds__(256)
void scan_fuse_kernel(const float* __restrict__ lc, const float* __restrict__ lz,
                      const float* __restrict__ lg, const float* __restrict__ xs,
                      __bf16* __restrict__ y, int T, int E)
{
    const int c = blockIdx.x * 256 + threadIdx.x;
    const int b = c / E;
    const int e = c - b * E;
    size_t idx = (size_t)b * T * E + e;

    float al = 0.0f;            // a_star (running cumsum of log_coeffs)
    float L  = -INFINITY;       // running logcumsumexp(log_values - a_star)
    for (int t = 0; t < T; ++t, idx += (size_t)E) {
        al += lc[idx];
        const float v  = lz[idx] + lg[idx] - al;
        const float mx = fmaxf(L, v);
        L = mx + __logf(__expf(L - mx) + __expf(v - mx));
        const float h = __expf(al + L);
        y[idx] = (__bf16)(xs[idx] + h);
    }
}

// ---------------------------------------------------------------------------
extern "C" void kernel_launch(void* const* d_in, const int* in_sizes, int n_in,
                              void* d_out, int out_size, void* d_ws, size_t ws_size,
                              hipStream_t stream) {
    constexpr int  Bq = 4, Tq = 2048, Dq = 1024, Eq = 4096;
    constexpr long MTOK = (long)Bq * Tq;              // 8192 tokens

    const float* x  = (const float*)d_in[0];
    const float* W1 = (const float*)d_in[1];
    const float* b1 = (const float*)d_in[2];
    const float* W2 = (const float*)d_in[3];
    const float* b2 = (const float*)d_in[4];
    const float* Wg = (const float*)d_in[5];
    const float* bg = (const float*)d_in[6];
    const float* Wo = (const float*)d_in[7];
    const float* bo = (const float*)d_in[8];
    float* out = (float*)d_out;

    char* ws = (char*)d_ws;
    size_t off = 0;
    __bf16* x_in   = (__bf16*)(ws + off); off += (size_t)MTOK * Eq * 2;      //  64 MB
    float*  x_skip = (float*) (ws + off); off += (size_t)MTOK * Eq * 4;      // 128 MB
    float*  lc     = (float*) (ws + off); off += (size_t)MTOK * Eq * 4;      // 128 MB
    float*  lz     = (float*) (ws + off); off += (size_t)MTOK * Eq * 4;      // 128 MB
    float*  lg     = (float*) (ws + off); off += (size_t)MTOK * Eq * 4;      // 128 MB
    __bf16* yb     = (__bf16*)(ws + off); off += (size_t)MTOK * Eq * 2;      //  64 MB
    __bf16* xb     = (__bf16*)(ws + off); off += (size_t)MTOK * Dq * 2;      //  16 MB
    __bf16* W1b    = (__bf16*)(ws + off); off += (size_t)Eq * Dq * 2;        //   8 MB
    __bf16* W2b    = (__bf16*)(ws + off); off += (size_t)Eq * Dq * 2;        //   8 MB
    __bf16* Wgb    = (__bf16*)(ws + off); off += (size_t)2 * Eq * Eq * 2;    //  64 MB
    __bf16* Wob    = (__bf16*)(ws + off); off += (size_t)Dq * Eq * 2;        //   8 MB

    const dim3 blk(256);

    // 0) one-time fp32 -> bf16 conversions (~312 MB traffic ~= 14 us @ 23.3 TB/s)
    cvt_f32_bf16_kernel<<<dim3((MTOK * Dq) / 1024), blk, 0, stream>>>(x, xb, MTOK * Dq);
    cvt_f32_bf16_kernel<<<dim3(((long)Eq * Dq) / 1024), blk, 0, stream>>>(W1, W1b, (long)Eq * Dq);
    cvt_f32_bf16_kernel<<<dim3(((long)Eq * Dq) / 1024), blk, 0, stream>>>(W2, W2b, (long)Eq * Dq);
    cvt_f32_bf16_kernel<<<dim3(((long)2 * Eq * Eq) / 1024), blk, 0, stream>>>(Wg, Wgb, (long)2 * Eq * Eq);
    cvt_f32_bf16_kernel<<<dim3(((long)Dq * Eq) / 1024), blk, 0, stream>>>(Wo, Wob, (long)Dq * Eq);

    // 1) x_in = silu(x @ W1^T + b1)   [8192 x 4096], K = 1024
    gemm_wmma_kernel<0><<<dim3(Eq / BN, MTOK / BM), blk, 0, stream>>>(
        xb, W1b, b1, x_in, nullptr, nullptr, (int)MTOK, Eq, Dq);

    // 2) x_skip = silu(x @ W2^T + b2)
    gemm_wmma_kernel<1><<<dim3(Eq / BN, MTOK / BM), blk, 0, stream>>>(
        xb, W2b, b2, x_skip, nullptr, nullptr, (int)MTOK, Eq, Dq);

    // 3) gate GEMM [8192 x 8192], K = 4096, epilogue -> lc / lz / lg
    gemm_wmma_kernel<2><<<dim3(2 * Eq / BN, MTOK / BM), blk, 0, stream>>>(
        x_in, Wgb, bg, lc, lz, lg, (int)MTOK, 2 * Eq, Eq);

    // 4) log-space scan over T, fused add + bf16 convert
    scan_fuse_kernel<<<dim3((Bq * Eq) / 256), blk, 0, stream>>>(
        lc, lz, lg, x_skip, yb, Tq, Eq);

    // 5) out = y @ Wo^T + bo   [8192 x 1024], K = 4096
    gemm_wmma_kernel<3><<<dim3(Dq / BN, MTOK / BM), blk, 0, stream>>>(
        yb, Wob, bo, out, nullptr, nullptr, (int)MTOK, Dq, Eq);
}
